// TinySelfAttention_60318520705680
// MI455X (gfx1250) — matrix operand order repaired and checked by reading, not compiled, not run on hardware
//
#include <hip/hip_runtime.h>

typedef __attribute__((ext_vector_type(16))) _Float16 v16h;
typedef __attribute__((ext_vector_type(8)))  _Float16 v8h;
typedef __attribute__((ext_vector_type(8)))  float    v8f;

#define WMMA_F16(a, b, c) \
  __builtin_amdgcn_wmma_f32_16x16x32_f16(false, (a), false, (b), (short)0, (c), false, false)

#if defined(__gfx1250__) && __has_builtin(__builtin_amdgcn_tensor_load_to_lds) && \
    __has_builtin(__builtin_amdgcn_s_wait_tensorcnt)
#define USE_TDM 1
typedef __attribute__((ext_vector_type(4))) unsigned int v4u;
typedef __attribute__((ext_vector_type(8))) int v8i;
typedef __attribute__((ext_vector_type(4))) int v4i;
#endif

// ---------------- wide fragment builders (2x ds_load_b128 each) ----------------
__device__ __forceinline__ v16h frag2(const _Float16* p0, const _Float16* p1) {
  const v8h lo = *reinterpret_cast<const v8h*>(p0);
  const v8h hi = *reinterpret_cast<const v8h*>(p1);
  return __builtin_shufflevector(lo, hi, 0, 1, 2, 3, 4, 5, 6, 7, 8, 9, 10, 11, 12, 13, 14, 15);
}

// A (16x32 row-major): VGPR v holds K = 2v + (v>=4 ? 8:0) + h*8  -> two contiguous 16B runs
__device__ __forceinline__ v16h load_A_frag(const _Float16* s, int lda, int lane) {
  const int m = lane & 15, h = lane >> 4;
  const _Float16* p = s + m * lda + h * 8;
  return frag2(p, p + 16);
}

// B (32x16) from n-major source (row = column n of B, col = k): same K order as A (h*8 + 0..7, 16 + h*8 + 0..7)
__device__ __forceinline__ v16h load_B_nmajor(const _Float16* s, int lda, int lane) {
  const int n = lane & 15, h = lane >> 4;
  const _Float16* p = s + n * lda + h * 8;
  return frag2(p, p + 16);
}

// ---------------- f32 -> f16 bulk convert (bandwidth-bound, done once) ----------------
__global__ void __launch_bounds__(256)
cvt_f32_f16(const float* __restrict__ s, _Float16* __restrict__ d, int n8) {
  const int i = blockIdx.x * 256 + threadIdx.x;
  if (i >= n8) return;
  const float4 f0 = reinterpret_cast<const float4*>(s)[i * 2];
  const float4 f1 = reinterpret_cast<const float4*>(s)[i * 2 + 1];
  v8h h;
  h[0] = (_Float16)f0.x; h[1] = (_Float16)f0.y; h[2] = (_Float16)f0.z; h[3] = (_Float16)f0.w;
  h[4] = (_Float16)f1.x; h[5] = (_Float16)f1.y; h[6] = (_Float16)f1.z; h[7] = (_Float16)f1.w;
  reinterpret_cast<v8h*>(d)[i] = h;
}

// ---------------- GEMM: Out[m,n] = sum_k A[m,k]*W[n,k] + bias[n] ----------------
// A, W already f16. M=16384, K=N=512. 256 threads / 8 waves, tile 128x64, K-step 32.
// Double-buffered LDS so global loads of step k+1 overlap WMMA of step k.
template <bool OUT_F32>
__global__ void __launch_bounds__(256)
gemm_xWT(const _Float16* __restrict__ A, const _Float16* __restrict__ W,
         const float* __restrict__ bias, void* __restrict__ Out) {
  __shared__ _Float16 At[2][128][40];
  __shared__ _Float16 Bt[2][64][40];

  const int t = threadIdx.x;
  const int wave = t >> 5, lane = t & 31;
  const int h = lane >> 4, nlo = lane & 15;
  const int m0 = blockIdx.y * 128;
  const int n0 = blockIdx.x * 64;

  const int arow = t >> 1, aseg = (t & 1) * 16;
  const int wrow = t >> 2, wseg = (t & 3) * 8;

  auto stage = [&](int k0, int buf) {
    const uint4* sa =
        reinterpret_cast<const uint4*>(A + (size_t)(m0 + arow) * 512 + k0 + aseg);
    *reinterpret_cast<uint4*>(&At[buf][arow][aseg]) = sa[0];
    *reinterpret_cast<uint4*>(&At[buf][arow][aseg + 8]) = sa[1];
    *reinterpret_cast<uint4*>(&Bt[buf][wrow][wseg]) =
        *reinterpret_cast<const uint4*>(W + (size_t)(n0 + wrow) * 512 + k0 + wseg);
  };

  v8f acc[4] = {};
  stage(0, 0);
  __syncthreads();

  for (int ks = 0; ks < 16; ++ks) {
    const int cur = ks & 1;
    if (ks + 1 < 16) stage((ks + 1) * 32, cur ^ 1);
    const v16h a = load_A_frag(&At[cur][wave * 16][0], 40, lane);
#pragma unroll
    for (int s = 0; s < 4; ++s) {
      const v16h b = load_B_nmajor(&Bt[cur][s * 16][0], 40, lane);
      acc[s] = WMMA_F16(a, b, acc[s]);
    }
    __syncthreads();
  }

#pragma unroll
  for (int s = 0; s < 4; ++s) {
    const int col = n0 + s * 16 + nlo;
    const float bv_ = bias[col];
#pragma unroll
    for (int v = 0; v < 8; ++v) {
      const int row = m0 + wave * 16 + v + h * 8;
      const float val = acc[s][v] + bv_;
      if (OUT_F32)
        ((float*)Out)[(size_t)row * 512 + col] = val;
      else
        ((_Float16*)Out)[(size_t)row * 512 + col] = (_Float16)val;
    }
  }
}

// ---------------- Flash attention ----------------
// 128 threads / 4 waves, 16 queries per block, 32-key chunks, D=512.
// K chunk staged by the Tensor Data Mover (padded rows, stride 520); V chunk staged
// transposed ([d][key], stride 40) so all WMMA fragments are 2x ds_load_b128.
#define KSTR 520  // 1040B row stride: 16B aligned, 260 dwords == 4 (mod 64) -> conflict-free
#define VTSTR 40

__global__ void __launch_bounds__(128)
attn_flash(const _Float16* __restrict__ Qh, const _Float16* __restrict__ Kh,
           const _Float16* __restrict__ Vh, const int* __restrict__ mask,
           _Float16* __restrict__ Oh) {
  __shared__ _Float16 Tt[512 * VTSTR];  // 40KB, time-shared: Q stage / K chunk / V^T chunk
  __shared__ _Float16 Pt[16 * 40];
  __shared__ float Sp[4 * 16 * 32];
  __shared__ float Sred[16 * 32];
  __shared__ float mrow[16], lrow[16], arowv[16];
  __shared__ int mk[32];

  const int t = threadIdx.x;
  const int wave = t >> 5, lane = t & 31;
  const int h = lane >> 4, nlo = lane & 15;
  const int b = blockIdx.y;
  const int q0 = blockIdx.x * 16;
  const size_t qbase = ((size_t)b * 2048 + q0) * 512;
  const size_t kbase = (size_t)b * 2048 * 512;
  const float scale = 0.04419417382415922f;  // 1/sqrt(512)

  // ---- stage Q (16 x 512) into Tt with stride KSTR ----
  {
    const int row = t >> 3, cs = (t & 7) * 64;
    const uint4* src = reinterpret_cast<const uint4*>(Qh + qbase + (size_t)row * 512 + cs);
#pragma unroll
    for (int j = 0; j < 8; ++j)
      *reinterpret_cast<uint4*>(&Tt[row * KSTR + cs + j * 8]) = src[j];
  }
  if (t < 16) { mrow[t] = -1e30f; lrow[t] = 0.0f; }
  __syncthreads();

  v16h qf[4];
#pragma unroll
  for (int c = 0; c < 4; ++c)
    qf[c] = load_A_frag(&Tt[wave * 128 + c * 32], KSTR, lane);
  v8f o[8] = {};
  __syncthreads();

  for (int kc = 0; kc < 2048; kc += 32) {
    // ---- phase 1: K chunk (32 x 512, stride 520) into Tt ----
#ifdef USE_TDM
    if (wave == 0) {
      const unsigned lds_off = (unsigned)(unsigned long long)(const void*)&Tt[0];
      const unsigned long long ga =
          (unsigned long long)(const void*)(Kh + kbase + (size_t)kc * 512);
      const v4u g0 = {1u,                                    // count=1
                      lds_off,                               // D#.lds_addr
                      (unsigned)ga,                          // D#.global_addr[31:0]
                      (unsigned)(ga >> 32) | (2u << 30)};    // addr[56:32] | type=2
      // data_size=2B | pad_enable | pad_interval=256dw(1024B) | pad_amount=4dw(16B)
      const v8i g1 = {(int)0x07D10000,
                      (int)(512u << 16),    // tensor_dim0 = 512
                      (int)(2048u << 16),   // tensor_dim1 = 2048
                      (int)(512u << 16),    // tile_dim0 = 512
                      32,                   // tile_dim1 = 32
                      512,                  // tensor_dim0_stride = 512
                      0, 0};
      const v4i gz4 = {0, 0, 0, 0};
      const v8i gz8 = {0, 0, 0, 0, 0, 0, 0, 0};
      __builtin_amdgcn_tensor_load_to_lds(g0, g1, gz4, gz4, gz8, 0);
      __builtin_amdgcn_s_wait_tensorcnt(0);
    }
#else
    {
      const int row = t >> 2, cs = (t & 3) * 128;
      const uint4* src =
          reinterpret_cast<const uint4*>(Kh + kbase + (size_t)(kc + row) * 512 + cs);
#pragma unroll
      for (int j = 0; j < 16; ++j)
        *reinterpret_cast<uint4*>(&Tt[row * KSTR + cs + j * 8]) = src[j];
    }
#endif
    if (t < 32) mk[t] = mask[b * 2048 + kc + t];
    __syncthreads();

    // ---- phase 2: partial S = Q_slice * K_slice^T; prefetch next chunk into L2 ----
    if (kc + 32 < 2048) {
      const int prow = t >> 2, pcs = (t & 3) * 128;
      const _Float16* pk = Kh + kbase + (size_t)(kc + 32 + prow) * 512 + pcs;
      const _Float16* pv = Vh + kbase + (size_t)(kc + 32 + prow) * 512 + pcs;
      __builtin_prefetch(pk, 0, 1);
      __builtin_prefetch(pk + 64, 0, 1);
      __builtin_prefetch(pv, 0, 1);
      __builtin_prefetch(pv + 64, 0, 1);
    }
    v8f sa[2] = {};
#pragma unroll
    for (int c = 0; c < 4; ++c) {
      const int dbase = wave * 128 + c * 32;
#pragma unroll
      for (int s = 0; s < 2; ++s) {
        const v16h bf = load_B_nmajor(&Tt[(s * 16) * KSTR + dbase], KSTR, lane);
        sa[s] = WMMA_F16(qf[c], bf, sa[s]);
      }
    }
#pragma unroll
    for (int s = 0; s < 2; ++s)
#pragma unroll
      for (int v = 0; v < 8; ++v)
        Sp[wave * 512 + (v + h * 8) * 32 + s * 16 + nlo] = sa[s][v];
    __syncthreads();

    // ---- phase 3: deterministic reduce of S + stage V^T (K reads are done) ----
#pragma unroll
    for (int j = 0; j < 4; ++j) {
      const int i = t * 4 + j;
      Sred[i] = Sp[i] + Sp[512 + i] + Sp[1024 + i] + Sp[1536 + i];
    }
    {
      const int kp = t & 15;           // key pair 2kp, 2kp+1
      const int cs = (t >> 4) * 64;    // 64-column segment
      const size_t r0 = kbase + (size_t)(kc + 2 * kp) * 512;
      unsigned* vtw = reinterpret_cast<unsigned*>(Tt);
#pragma unroll
      for (int j = 0; j < 8; ++j) {
        const int c = cs + j * 8;
        const uint4 a = *reinterpret_cast<const uint4*>(Vh + r0 + c);
        const uint4 bb = *reinterpret_cast<const uint4*>(Vh + r0 + 512 + c);
        const unsigned aw[4] = {a.x, a.y, a.z, a.w};
        const unsigned bw[4] = {bb.x, bb.y, bb.z, bb.w};
#pragma unroll
        for (int e = 0; e < 4; ++e) {
          const unsigned w0 = (aw[e] & 0xFFFFu) | (bw[e] << 16);
          const unsigned w1 = (aw[e] >> 16) | (bw[e] & 0xFFFF0000u);
          vtw[(c + 2 * e) * (VTSTR / 2) + kp] = w0;
          vtw[(c + 2 * e + 1) * (VTSTR / 2) + kp] = w1;
        }
      }
    }
    __syncthreads();

    // ---- phase 4: online softmax (16 rows on first 16 lanes) ----
    if (t < 16) {
      const float mo = mrow[t];
      float mx = mo;
      float sv[32];
#pragma unroll
      for (int k = 0; k < 32; ++k) {
        const float s_ = mk[k] ? Sred[t * 32 + k] * scale : -10000.0f;
        sv[k] = s_;
        mx = fmaxf(mx, s_);
      }
      const float al = __expf(mo - mx);
      float sum = 0.0f;
#pragma unroll
      for (int k = 0; k < 32; ++k) {
        const float p = __expf(sv[k] - mx);
        sum += p;
        Pt[t * 40 + k] = (_Float16)p;
      }
      mrow[t] = mx;
      lrow[t] = lrow[t] * al + sum;
      arowv[t] = al;
    }
    __syncthreads();

    // ---- phase 5: rescale running O, then O += P * V_slice ----
    float av[8];
#pragma unroll
    for (int v = 0; v < 8; ++v) av[v] = arowv[h * 8 + v];
#pragma unroll
    for (int s = 0; s < 8; ++s)
#pragma unroll
      for (int v = 0; v < 8; ++v) o[s][v] *= av[v];

    const v16h pf = load_A_frag(Pt, 40, lane);
#pragma unroll
    for (int s = 0; s < 8; ++s) {
      const v16h bf = load_B_nmajor(&Tt[(size_t)(wave * 128 + s * 16) * VTSTR], VTSTR, lane);
      o[s] = WMMA_F16(pf, bf, o[s]);
    }
    __syncthreads();
  }

  // ---- epilogue ----
  float li[8];
#pragma unroll
  for (int v = 0; v < 8; ++v) li[v] = 1.0f / lrow[h * 8 + v];
#pragma unroll
  for (int s = 0; s < 8; ++s) {
    const int col = wave * 128 + s * 16 + nlo;
#pragma unroll
    for (int v = 0; v < 8; ++v) {
      const int row = v + h * 8;
      Oh[qbase + (size_t)row * 512 + col] = (_Float16)(o[s][v] * li[v]);
    }
  }
}

// ---------------- host ----------------
extern "C" void kernel_launch(void* const* d_in, const int* in_sizes, int n_in,
                              void* d_out, int out_size, void* d_ws, size_t ws_size,
                              hipStream_t stream) {
  (void)in_sizes; (void)n_in; (void)out_size; (void)ws_size;

  const float* x    = (const float*)d_in[0];
  const int*   mask = (const int*)d_in[1];
  const float* Wq   = (const float*)d_in[2];
  const float* bq   = (const float*)d_in[3];
  const float* Wk   = (const float*)d_in[4];
  const float* bk   = (const float*)d_in[5];
  const float* Wv   = (const float*)d_in[6];
  const float* bv   = (const float*)d_in[7];
  const float* Wp   = (const float*)d_in[8];
  const float* bp   = (const float*)d_in[9];
  float* out = (float*)d_out;

  const size_t NE = (size_t)8 * 2048 * 512;  // 8.4M tokens*features
  const size_t WE = (size_t)512 * 512;
  _Float16* xh  = (_Float16*)d_ws;
  _Float16* Whq = xh + NE;
  _Float16* Whk = Whq + WE;
  _Float16* Whv = Whk + WE;
  _Float16* Whp = Whv + WE;
  _Float16* Qh  = Whp + WE;
  _Float16* Kh  = Qh + NE;
  _Float16* Vh  = Kh + NE;
  _Float16* Oh  = Vh + NE;

  // one-time f32->f16 conversions (x and weights)
  cvt_f32_f16<<<(int)(NE / 8 / 256), 256, 0, stream>>>(x, xh, (int)(NE / 8));
  cvt_f32_f16<<<(int)(WE / 8 / 256), 256, 0, stream>>>(Wq, Whq, (int)(WE / 8));
  cvt_f32_f16<<<(int)(WE / 8 / 256), 256, 0, stream>>>(Wk, Whk, (int)(WE / 8));
  cvt_f32_f16<<<(int)(WE / 8 / 256), 256, 0, stream>>>(Wv, Whv, (int)(WE / 8));
  cvt_f32_f16<<<(int)(WE / 8 / 256), 256, 0, stream>>>(Wp, Whp, (int)(WE / 8));

  const dim3 gg(8, 128);
  const dim3 gb(256);
  gemm_xWT<false><<<gg, gb, 0, stream>>>(xh, Whq, bq, Qh);
  gemm_xWT<false><<<gg, gb, 0, stream>>>(xh, Whk, bk, Kh);
  gemm_xWT<false><<<gg, gb, 0, stream>>>(xh, Whv, bv, Vh);

  attn_flash<<<dim3(128, 8), 128, 0, stream>>>(Qh, Kh, Vh, mask, Oh);

  gemm_xWT<true><<<gg, gb, 0, stream>>>(Oh, Whp, bp, out);
}
